// PriorFocalModifierLoss_40956808134749
// MI455X (gfx1250) — matrix operand order repaired and checked by reading, not compiled, not run on hardware
//
#include <hip/hip_runtime.h>

#define B_DIM 128
#define C_DIM 8192

#define GAMMA_NEG       3.0f
#define GAMMA_POS       1.0f
#define GAMMA_CLASS_NG  1.2f
#define GAMMA_CLASS_POS 1.0f
#define CLIP_C          0.05f
#define EPS_C           1e-8f

typedef __attribute__((ext_vector_type(16))) __bf16 v16bf;
typedef __attribute__((ext_vector_type(8)))  __bf16 v8bf;
typedef __attribute__((ext_vector_type(2)))  __bf16 v2bf;
typedef __attribute__((ext_vector_type(8)))  float  v8f;

// f32 -> bf16 (RNE) for two floats, packed into one dword.
// Expressed as native fptrunc so the backend can select v_cvt_pk_bf16_f32.
__device__ __forceinline__ unsigned pack_bf16x2(float lo, float hi) {
    v2bf v = { (__bf16)lo, (__bf16)hi };
    return __builtin_bit_cast(unsigned, v);
}

// ---------------------------------------------------------------------------
// Kernel 1: att_raw = y @ co via v_wmma_f32_16x16x32_bf16
//   grid = (C/256, B/16), block = 256 (8 waves, wave32).
//   WG tile: M=16, N=256, K-step=32; wave owns two 16x16 N-tiles (A reused).
//   co tile stored TRANSPOSED in LDS as packed bf16-pairs along K:
//   staging = 16x ds_store_b32, B fragments = 2x ds_load_b128.
//   Software-pipelined, double-buffered, 1 barrier per K-step, literal
//   buffer indices, branch-free steady-state (tail peeled).
// ---------------------------------------------------------------------------
__global__ __launch_bounds__(256)
void att_wmma_kernel(const float* __restrict__ y,
                     const float* __restrict__ co,
                     float* __restrict__ att_raw) {
    // row stride 40 bf16 = 80 B -> every row 16-B aligned for b128 LDS loads
    __shared__ __align__(16) __bf16 As[2][16][40];    // y tile  (M x K)
    __shared__ __align__(16) __bf16 Bt[2][256][40];   // co tile (N x K), transposed

    const int t    = threadIdx.x;
    const int lane = t & 31;
    const int wave = t >> 5;
    const int m0   = blockIdx.y * 16;
    const int n0   = blockIdx.x * 256;
    const int col  = lane & 15;     // M-row (A) / N-col (B)
    const int hi   = lane >> 4;     // half-wave select

    // co staging: thread t covers c4 = t&63 (N/4), base k-pair row = t>>6,
    // j in 0..3 adds 4 pair-rows -> 16 pair-rows = 32 K values.
    const int tt6 = t >> 6;          // 0..3
    const int nn  = (t & 63) * 4;    // N within tile

    // y staging: thread t covers row t>>4 (M), cols (t&15)*2 .. +1
    const int yr_row = t >> 4;
    const int yr_col = (t & 15) * 2;

    const float* co_ptr = co + (size_t)(2 * tt6) * C_DIM + n0 + nn;
    const float* y_ptr  = y  + (size_t)(m0 + yr_row) * C_DIM + yr_col;

    float4 cr0[4], cr1[4];           // two adjacent K rows per j
    float2 yr;

    auto load_tile = [&]() {
#pragma unroll
        for (int j = 0; j < 4; ++j) {
            // k rows: 2*(tt6 + 4j) and +1 ; byte offsets are immediates
            cr0[j] = *reinterpret_cast<const float4*>(co_ptr + (size_t)(8 * j) * C_DIM);
            cr1[j] = *reinterpret_cast<const float4*>(co_ptr + (size_t)(8 * j + 1) * C_DIM);
        }
        yr = *reinterpret_cast<const float2*>(y_ptr);
        co_ptr += (size_t)32 * C_DIM;   // advance one K-step
        y_ptr  += 32;
    };

    auto store_tile = [&](int p) {
        *reinterpret_cast<unsigned*>(&As[p][yr_row][yr_col]) = pack_bf16x2(yr.x, yr.y);
#pragma unroll
        for (int j = 0; j < 4; ++j) {
            const int kk = 2 * (tt6 + 4 * j);   // even K index
            *reinterpret_cast<unsigned*>(&Bt[p][nn + 0][kk]) = pack_bf16x2(cr0[j].x, cr1[j].x);
            *reinterpret_cast<unsigned*>(&Bt[p][nn + 1][kk]) = pack_bf16x2(cr0[j].y, cr1[j].y);
            *reinterpret_cast<unsigned*>(&Bt[p][nn + 2][kk]) = pack_bf16x2(cr0[j].z, cr1[j].z);
            *reinterpret_cast<unsigned*>(&Bt[p][nn + 3][kk]) = pack_bf16x2(cr0[j].w, cr1[j].w);
        }
    };

    v8f acc0 = {}, acc1 = {};

    auto compute = [&](int p) {
        // A fragment (16x32 bf16): elems 0..7 -> K = hi*16+0..7,
        //                          elems 8..15 -> K = hi*16+8..15
        const v8bf alo = *reinterpret_cast<const v8bf*>(&As[p][col][hi * 16]);
        const v8bf ahi = *reinterpret_cast<const v8bf*>(&As[p][col][hi * 16 + 8]);
        const v16bf a = __builtin_shufflevector(alo, ahi,
            0,1,2,3,4,5,6,7,8,9,10,11,12,13,14,15);

        // B fragments (32x16 bf16): lanes 0-15 K=0..15, lanes 16-31 K=16..31
        const int nb0 = wave * 32 + col;
        const v8bf b0lo = *reinterpret_cast<const v8bf*>(&Bt[p][nb0][hi * 16 + 0]);
        const v8bf b0hi = *reinterpret_cast<const v8bf*>(&Bt[p][nb0][hi * 16 + 8]);
        const v16bf b0 = __builtin_shufflevector(b0lo, b0hi,
            0,1,2,3,4,5,6,7,8,9,10,11,12,13,14,15);

        const v8bf b1lo = *reinterpret_cast<const v8bf*>(&Bt[p][nb0 + 16][hi * 16 + 0]);
        const v8bf b1hi = *reinterpret_cast<const v8bf*>(&Bt[p][nb0 + 16][hi * 16 + 8]);
        const v16bf b1 = __builtin_shufflevector(b1lo, b1hi,
            0,1,2,3,4,5,6,7,8,9,10,11,12,13,14,15);

        acc0 = __builtin_amdgcn_wmma_f32_16x16x32_bf16(
            false, a, false, b0, (short)0, acc0, false, false);
        acc1 = __builtin_amdgcn_wmma_f32_16x16x32_bf16(
            false, a, false, b1, (short)0, acc1, false, false);
    };

    const int NSTEP = C_DIM / 32;   // 256 (even)

    load_tile();              // k-step 0
    store_tile(0);
    __syncthreads();

    // branch-free steady state: two K-steps per iteration
    for (int s = 0; s < NSTEP - 2; s += 2) {
        load_tile();          // k-step s+1
        compute(0);
        store_tile(1);
        __syncthreads();

        load_tile();          // k-step s+2
        compute(1);
        store_tile(0);
        __syncthreads();
    }
    // tail: k-steps NSTEP-2, NSTEP-1
    load_tile();              // k-step NSTEP-1 (last)
    compute(0);
    store_tile(1);
    __syncthreads();
    compute(1);

    // C/D layout: VGPR r -> M = r + 8*hi, N = col
#pragma unroll
    for (int r = 0; r < 8; ++r) {
        const size_t row = (size_t)(m0 + r + hi * 8) * C_DIM + n0 + wave * 32 + col;
        att_raw[row]      = acc0[r];
        att_raw[row + 16] = acc1[r];
    }
}

// ---------------------------------------------------------------------------
// Kernel 2: row sums of att_raw (normalization denominator)
// ---------------------------------------------------------------------------
__global__ __launch_bounds__(256)
void rowsum_kernel(const float* __restrict__ att_raw, float* __restrict__ rowsum) {
    __shared__ float s[256];
    const int b = blockIdx.x;
    float acc = 0.f;
    for (int c = threadIdx.x; c < C_DIM; c += 256)
        acc += att_raw[(size_t)b * C_DIM + c];
    s[threadIdx.x] = acc;
    __syncthreads();
    for (int o = 128; o > 0; o >>= 1) {
        if (threadIdx.x < o) s[threadIdx.x] += s[threadIdx.x + o];
        __syncthreads();
    }
    if (threadIdx.x == 0) rowsum[b] = s[0];
}

// ---------------------------------------------------------------------------
// Kernel 3: elementwise focal loss -> per-block partial sums (deterministic)
// ---------------------------------------------------------------------------
__global__ __launch_bounds__(256)
void loss_kernel(const float* __restrict__ x, const float* __restrict__ y,
                 const float* __restrict__ att_raw, const float* __restrict__ rowsum,
                 const float* __restrict__ weight, float* __restrict__ partials) {
    __shared__ float s[256];
    const int gid = blockIdx.x * 256 + threadIdx.x;   // [0, B*C)
    const int b = gid >> 13;                           // / 8192
    const int c = gid & (C_DIM - 1);

    const float raw = att_raw[gid];
    const float att = raw / rowsum[b];
    const float xv  = x[gid];
    const float yv  = y[gid];
    const float w   = weight[c];

    const float x_sig  = 1.0f / (1.0f + __expf(-xv));
    const float xs_pos = x_sig * (GAMMA_CLASS_POS - att);
    float xs_neg = fminf(1.0f - x_sig + CLIP_C, 1.0f);
    xs_neg = fminf((raw == 0.0f) ? xs_neg : xs_neg * GAMMA_CLASS_NG, 1.0f);

    const float los_pos = yv * __logf(fmaxf(xs_pos, EPS_C));
    const float los_neg = (1.0f - yv) * __logf(fmaxf(xs_neg, EPS_C));
    const float loss    = (los_pos + los_neg) * w;

    const float pt  = xs_pos * yv + xs_neg * (1.0f - yv);
    const float osg = GAMMA_POS * yv + (GAMMA_NEG + w) * (1.0f - yv);
    const float osw = __powf(fmaxf(1.0f - pt, 0.0f), osg);

    s[threadIdx.x] = -(loss * osw);
    __syncthreads();
    for (int o = 128; o > 0; o >>= 1) {
        if (threadIdx.x < o) s[threadIdx.x] += s[threadIdx.x + o];
        __syncthreads();
    }
    if (threadIdx.x == 0) partials[blockIdx.x] = s[0];
}

// ---------------------------------------------------------------------------
// Kernel 4: final deterministic reduction of block partials
// ---------------------------------------------------------------------------
__global__ __launch_bounds__(256)
void final_reduce_kernel(const float* __restrict__ partials,
                         float* __restrict__ out, int n) {
    __shared__ float s[256];
    float acc = 0.f;
    for (int i = threadIdx.x; i < n; i += 256) acc += partials[i];
    s[threadIdx.x] = acc;
    __syncthreads();
    for (int o = 128; o > 0; o >>= 1) {
        if (threadIdx.x < o) s[threadIdx.x] += s[threadIdx.x + o];
        __syncthreads();
    }
    if (threadIdx.x == 0) out[0] = s[0];
}

extern "C" void kernel_launch(void* const* d_in, const int* in_sizes, int n_in,
                              void* d_out, int out_size, void* d_ws, size_t ws_size,
                              hipStream_t stream) {
    const float* x  = (const float*)d_in[0];
    const float* y  = (const float*)d_in[1];
    const float* co = (const float*)d_in[2];
    const float* w  = (const float*)d_in[3];

    float* ws       = (float*)d_ws;
    float* att_raw  = ws;                                   // B*C floats
    float* rowsum   = att_raw + (size_t)B_DIM * C_DIM;      // B floats
    float* partials = rowsum + B_DIM;                       // B*C/256 floats

    const int n_loss_blocks = (B_DIM * C_DIM) / 256;        // 4096

    dim3 g1(C_DIM / 256, B_DIM / 16);                       // (32, 8)
    att_wmma_kernel<<<g1, 256, 0, stream>>>(y, co, att_raw);
    rowsum_kernel<<<B_DIM, 256, 0, stream>>>(att_raw, rowsum);
    loss_kernel<<<n_loss_blocks, 256, 0, stream>>>(x, y, att_raw, rowsum, w, partials);
    final_reduce_kernel<<<1, 256, 0, stream>>>(partials, (float*)d_out, n_loss_blocks);
}
